// SelfAttention_47467978555690
// MI455X (gfx1250) — matrix operand order repaired and checked by reading, not compiled, not run on hardware
//
#include <hip/hip_runtime.h>
#include <hip/hip_bf16.h>

typedef _Float16 f16_t;
typedef __attribute__((ext_vector_type(16))) _Float16 v16h;
typedef __attribute__((ext_vector_type(8)))  _Float16 v8h;
typedef __attribute__((ext_vector_type(8)))  float    v8f;
typedef __attribute__((ext_vector_type(4)))  unsigned v4u;
typedef __attribute__((ext_vector_type(8)))  unsigned v8u;
typedef __attribute__((ext_vector_type(4)))  int      v4i_t;
typedef __attribute__((ext_vector_type(8)))  int      v8i_t;

#define NB   32           // batch
#define HW   1024         // tokens per batch
#define CH   512          // channels / head dim
#define GRP  8
#define CG   (CH / GRP)   // 64
#define MTOT (NB * HW)    // 32768 rows
#define EPS  1e-3f

#if defined(__has_builtin)
#if __has_builtin(__builtin_amdgcn_tensor_load_to_lds) && \
    __has_builtin(__builtin_amdgcn_s_wait_tensorcnt)
#define HAVE_TDM 1
#endif
#endif
#ifndef HAVE_TDM
#define HAVE_TDM 0
#endif

// LDS tile rows: 32 data halves (64B) + TDM pad of 4 dwords (16B) = 40 halves.
// Bank stride 80B/4 = 20 banks: 20*r mod 64 distinct for r=0..15 -> clean rows.
#define ROWP 40

#if HAVE_TDM
// ---------------------------------------------------------------------------
// Issue one TDM 2-D tile load: global (rows x 32 halves, row stride `stride`
// halves) -> LDS at byte offset lds_addr, padded 16B per 64B row.
// Descriptor packing per CDNA5 ISA section 8 (D# groups). data_size=1 (2B),
// type=2 (image), count=1, pad_enable=1, pad_interval=3 (16 dw), pad_amount=3
// (4 dw). Tile treated as its own tensor (tensor_dim == tile_dim): no OOB.
// ---------------------------------------------------------------------------
__device__ __forceinline__ void tdm_load_2d(unsigned lds_addr, const void* gp,
                                            unsigned stride, unsigned rows) {
  const unsigned long long ga = (unsigned long long)(size_t)gp;
  v4u g0;
  g0[0] = 1u;                                               // count=1
  g0[1] = lds_addr;                                         // lds_addr[31:0]
  g0[2] = (unsigned)ga;                                     // global_addr[31:0]
  g0[3] = (unsigned)((ga >> 32) & 0x1FFFFFFu) | (2u << 30); // ga[56:32] | type=2
  v8u g1;
  g1[0] = (1u << 16) | (1u << 20) | (3u << 22) | (3u << 25); // 2B | pad 4dw/16dw
  g1[1] = (32u & 0xFFFFu) << 16;                 // tensor_dim0[15:0]=32
  g1[2] = (rows & 0xFFFFu) << 16;                // td0 hi=0 | tensor_dim1 lo
  g1[3] = (32u << 16);                           // td1 hi=0 | tile_dim0=32
  g1[4] = rows;                                  // tile_dim1=rows, tile_dim2=0
  g1[5] = stride;                                // tensor_dim0_stride lo
  g1[6] = 0u;                                    // stride0 hi | stride1 lo
  g1[7] = 0u;                                    // stride1 hi
  const v4i_t z4 = {};
#if __clang_major__ >= 23
  const v8i_t z8 = {};
  __builtin_amdgcn_tensor_load_to_lds(g0, __builtin_bit_cast(v8i_t, g1),
                                      z4, z4, z8, 0);
#else
  __builtin_amdgcn_tensor_load_to_lds(g0, __builtin_bit_cast(v8i_t, g1),
                                      z4, z4, 0);
#endif
}
#endif  // HAVE_TDM

// ---------------------------------------------------------------------------
// Fragment packing for v_wmma_f32_16x16x32_f16 (wave32), per ISA 7.12.2:
// A: lane r=lane&15 holds row r; halves K = {ko..ko+7, 16+ko..16+ko+7},
//    ko = 8*(lane>=16)  -> two 16B loads at +0 and +16 halves.
// B: lane n=lane&15 holds col n; halves K = {ko..ko+7, 16+ko..16+ko+7},
//    ko = 8*(lane>=16)  -> two 16B loads at +0 and +16 halves.
// D: v8f elem i -> (m = 8*(lane>=16)+i, n = lane&15).
// ---------------------------------------------------------------------------
__device__ __forceinline__ v16h pack_a(const f16_t* p) {
  v8h lo = *(const v8h*)(p);
  v8h hi = *(const v8h*)(p + 16);
  v16h a;
#pragma unroll
  for (int i = 0; i < 8; ++i) { a[i] = lo[i]; a[i + 8] = hi[i]; }
  return a;
}
__device__ __forceinline__ v16h pack_b(const f16_t* p) {
  v8h lo = *(const v8h*)(p);
  v8h hi = *(const v8h*)(p + 16);
  v16h b;
#pragma unroll
  for (int i = 0; i < 8; ++i) { b[i] = lo[i]; b[i + 8] = hi[i]; }
  return b;
}

// ---------------------------------------------------------------------------
// TDM-fed, double-buffered LDS GEMM block tile.
// Block = 256 thr = 8 waves in 4(m) x 2(n); wave tile 32x32 (2x2 WMMA accs);
// block tile 128(m) x 64(n), K step 32.
// Ablk: row-major [128 x K] at block m-origin; Bblk: column-major [64 x K].
// Pipeline per K-step: wave0 DMAs next A/B tiles (TENSORcnt), barrier,
// compute current from LDS, barrier.
// ---------------------------------------------------------------------------
__device__ __forceinline__ void wmma_gemm_tile(const f16_t* __restrict__ Ablk, int lda,
                                               const f16_t* __restrict__ Bblk, int ldb,
                                               int K, v8f acc[2][2]) {
  __shared__ f16_t As[2][128][ROWP];
  __shared__ f16_t Bs[2][64][ROWP];
  const int tid  = threadIdx.x;
  const int wave = tid >> 5, lane = tid & 31;
  const int mo = (wave >> 1) * 32, no = (wave & 1) * 32;
  const int r = lane & 15, hl = lane >> 4;
  const int NT = K >> 5;

#if HAVE_TDM
#define FETCH(buf, k0)                                                        \
  do {                                                                        \
    if (wave == 0) {                                                          \
      tdm_load_2d((unsigned)(size_t)&As[buf][0][0], Ablk + (k0), lda, 128);   \
      tdm_load_2d((unsigned)(size_t)&Bs[buf][0][0], Bblk + (k0), ldb, 64);    \
    }                                                                         \
  } while (0)
// s_wait_tensorcnt requires a constant-integer immediate: branch to constants.
#define WAIT_FETCH(more)                                                      \
  do {                                                                        \
    if (wave == 0) {                                                          \
      if (more) __builtin_amdgcn_s_wait_tensorcnt((short)2);                  \
      else      __builtin_amdgcn_s_wait_tensorcnt((short)0);                  \
    }                                                                         \
  } while (0)
#else
  const int rowA = tid >> 2, qq = (tid & 3) * 8;   // cooperative sync copy
#define FETCH(buf, k0)                                                        \
  do {                                                                        \
    *(v8h*)&As[buf][rowA][qq] =                                               \
        *(const v8h*)(Ablk + (size_t)rowA * lda + (k0) + qq);                 \
    *(v8h*)&As[buf][rowA + 64][qq] =                                          \
        *(const v8h*)(Ablk + (size_t)(rowA + 64) * lda + (k0) + qq);          \
    *(v8h*)&Bs[buf][rowA][qq] =                                               \
        *(const v8h*)(Bblk + (size_t)rowA * ldb + (k0) + qq);                 \
  } while (0)
#define WAIT_FETCH(more) do { } while (0)
#endif

  FETCH(0, 0);
  for (int kt = 0; kt < NT; ++kt) {
    const int cur = kt & 1;
    if (kt + 1 < NT) FETCH(cur ^ 1, (kt + 1) * 32);
    WAIT_FETCH(kt + 1 < NT);             // in-order TENSORcnt: cur's 2 done
    __syncthreads();                     // cur tile visible to all waves
    const v16h a0 = pack_a(&As[cur][mo + r][hl * 8]);
    const v16h a1 = pack_a(&As[cur][mo + r + 16][hl * 8]);
    const v16h b0 = pack_b(&Bs[cur][no + r][hl * 8]);
    const v16h b1 = pack_b(&Bs[cur][no + r + 16][hl * 8]);
    acc[0][0] = __builtin_amdgcn_wmma_f32_16x16x32_f16(false, a0, false, b0, (short)0, acc[0][0], false, false);
    acc[0][1] = __builtin_amdgcn_wmma_f32_16x16x32_f16(false, a0, false, b1, (short)0, acc[0][1], false, false);
    acc[1][0] = __builtin_amdgcn_wmma_f32_16x16x32_f16(false, a1, false, b0, (short)0, acc[1][0], false, false);
    acc[1][1] = __builtin_amdgcn_wmma_f32_16x16x32_f16(false, a1, false, b1, (short)0, acc[1][1], false, false);
    __syncthreads();                     // all reads of cur done before reuse
  }
#undef FETCH
#undef WAIT_FETCH
}

// Wave-tile origin for epilogues (must match wmma_gemm_tile's internal grid).
__device__ __forceinline__ void tile_origin(int bx, int by, int& m0, int& n0) {
  const int wave = threadIdx.x >> 5;
  m0 = bx * 128 + (wave >> 1) * 32;
  n0 = by * 64  + (wave & 1)  * 32;
}

// ---------------------------------------------------------------------------
// 1) GroupNorm statistics: one block per (b,g); reduce 1024*64 elements.
// ---------------------------------------------------------------------------
__global__ void k_gn_stats(const float* __restrict__ x, float* __restrict__ stats) {
  const int bg = blockIdx.x;
  const int b = bg >> 3, g = bg & 7;
  const float* xb = x + (size_t)b * HW * CH + g * CG;
  float s = 0.f, ss = 0.f;
  for (int i = threadIdx.x; i < HW * CG; i += 256) {
    float v = xb[(size_t)(i >> 6) * CH + (i & 63)];
    s += v; ss += v * v;
  }
#pragma unroll
  for (int off = 16; off > 0; off >>= 1) {
    s  += __shfl_down(s,  off, 32);
    ss += __shfl_down(ss, off, 32);
  }
  __shared__ float rs[8], rss[8];
  const int lane = threadIdx.x & 31, wv = threadIdx.x >> 5;
  if (lane == 0) { rs[wv] = s; rss[wv] = ss; }
  __syncthreads();
  if (threadIdx.x == 0) {
    float ts = 0.f, tss = 0.f;
#pragma unroll
    for (int i = 0; i < 8; ++i) { ts += rs[i]; tss += rss[i]; }
    const float inv  = 1.0f / (float)(HW * CG);
    const float mean = ts * inv;
    const float var  = tss * inv - mean * mean;
    stats[bg * 2 + 0] = mean;
    stats[bg * 2 + 1] = rsqrtf(var + EPS);
  }
}

// ---------------------------------------------------------------------------
// 2) Transpose + fp32->f16: src[K][N] row-major -> dst[N][K] (column-major B).
// ---------------------------------------------------------------------------
__global__ void k_wt_cvt(const float* __restrict__ src, f16_t* __restrict__ dst,
                         int K, int N) {
  const int idx = blockIdx.x * blockDim.x + threadIdx.x;
  if (idx >= K * N) return;
  const int k = idx / N, n = idx % N;
  dst[(size_t)n * K + k] = (f16_t)src[idx];
}

// ---------------------------------------------------------------------------
// 3) Apply GroupNorm -> h (f16), vectorized 4 channels/thread.
// ---------------------------------------------------------------------------
__global__ void k_gn_apply(const float* __restrict__ x, const float* __restrict__ gamma,
                           const float* __restrict__ beta, const float* __restrict__ stats,
                           f16_t* __restrict__ h) {
  const int idx = (blockIdx.x * blockDim.x + threadIdx.x) * 4;   // < 16,777,216
  const int c = idx & (CH - 1);
  const int b = idx >> 19;                                       // / (HW*CH)
  const int g = c >> 6;                                          // 4 chans share a group
  const float mean = stats[(b * GRP + g) * 2 + 0];
  const float rstd = stats[(b * GRP + g) * 2 + 1];
  const float4 xv = *(const float4*)(x + (size_t)idx);
#pragma unroll
  for (int j = 0; j < 4; ++j) {
    const float xj = (&xv.x)[j];
    h[(size_t)idx + j] = (f16_t)((xj - mean) * rstd * gamma[c + j] + beta[c + j]);
  }
}

// ---------------------------------------------------------------------------
// 4) QKV GEMM: [32768, 512] x wT[1536][512] (+bias) -> q,k (row-major f16),
//    v stored TRANSPOSED as vT[b][c][token] so attn*V has column-major B.
// ---------------------------------------------------------------------------
__global__ void k_qkv(const f16_t* __restrict__ h, const f16_t* __restrict__ wT,
                      const float* __restrict__ bias,
                      f16_t* __restrict__ q, f16_t* __restrict__ kmat,
                      f16_t* __restrict__ vT) {
  v8f acc[2][2] = {};
  wmma_gemm_tile(h + (size_t)blockIdx.x * 128 * CH, CH,
                 wT + (size_t)blockIdx.y * 64 * CH, CH, CH, acc);
  int m0, n0;
  tile_origin(blockIdx.x, blockIdx.y, m0, n0);
  const int lane = threadIdx.x & 31;
#pragma unroll
  for (int mi = 0; mi < 2; ++mi)
#pragma unroll
    for (int ni = 0; ni < 2; ++ni) {
      const int n = n0 + ni * 16 + (lane & 15);
      const int mb = m0 + mi * 16 + (lane >> 4) * 8;
      const float bn = bias[n];
#pragma unroll
      for (int i = 0; i < 8; ++i) {
        const int m = mb + i;
        const int b = m >> 10, tok = m & 1023;
        const f16_t val = (f16_t)(acc[mi][ni][i] + bn);
        if (n < CH)           q[((size_t)b * HW + tok) * CH + n]             = val;
        else if (n < 2 * CH)  kmat[((size_t)b * HW + tok) * CH + (n - CH)]   = val;
        else                  vT[((size_t)b * CH + (n - 2 * CH)) * HW + tok] = val;
      }
    }
}

// ---------------------------------------------------------------------------
// 5) Scores: S[b] = Q[b] (1024x512) @ K[b]^T, fp32 out (scale in softmax).
//    K row-major [token][C] == column-major B operand for free.
// ---------------------------------------------------------------------------
__global__ void k_scores(const f16_t* __restrict__ q, const f16_t* __restrict__ kmat,
                         float* __restrict__ S) {
  const int b = blockIdx.z;
  v8f acc[2][2] = {};
  wmma_gemm_tile(q    + (size_t)b * HW * CH + (size_t)blockIdx.x * 128 * CH, CH,
                 kmat + (size_t)b * HW * CH + (size_t)blockIdx.y * 64  * CH, CH, CH, acc);
  float* Sb = S + (size_t)b * HW * HW;
  int m0, n0;
  tile_origin(blockIdx.x, blockIdx.y, m0, n0);
  const int lane = threadIdx.x & 31;
#pragma unroll
  for (int mi = 0; mi < 2; ++mi)
#pragma unroll
    for (int ni = 0; ni < 2; ++ni) {
      const int n = n0 + ni * 16 + (lane & 15);
      const int mb = m0 + mi * 16 + (lane >> 4) * 8;
#pragma unroll
      for (int i = 0; i < 8; ++i) Sb[(size_t)(mb + i) * HW + n] = acc[mi][ni][i];
    }
}

// ---------------------------------------------------------------------------
// 6) Row softmax with 1/sqrt(512) scale; fp32 math, f16 probabilities out.
// ---------------------------------------------------------------------------
__global__ void k_softmax(const float* __restrict__ S, f16_t* __restrict__ P) {
  const int row = blockIdx.x, b = blockIdx.y;
  const float scale = 0.04419417382415922f;  // 1/sqrt(512)
  const float* s = S + ((size_t)b * HW + row) * HW;
  f16_t*       p = P + ((size_t)b * HW + row) * HW;
  __shared__ float red[8];
  float v[4];
  float m = -3.4e38f;
#pragma unroll
  for (int j = 0; j < 4; ++j) {
    v[j] = s[threadIdx.x + j * 256] * scale;
    m = fmaxf(m, v[j]);
  }
#pragma unroll
  for (int off = 16; off > 0; off >>= 1) m = fmaxf(m, __shfl_down(m, off, 32));
  const int lane = threadIdx.x & 31, wv = threadIdx.x >> 5;
  if (lane == 0) red[wv] = m;
  __syncthreads();
  m = fmaxf(fmaxf(fmaxf(red[0], red[1]), fmaxf(red[2], red[3])),
            fmaxf(fmaxf(red[4], red[5]), fmaxf(red[6], red[7])));
  __syncthreads();
  float sum = 0.f;
#pragma unroll
  for (int j = 0; j < 4; ++j) { v[j] = expf(v[j] - m); sum += v[j]; }
#pragma unroll
  for (int off = 16; off > 0; off >>= 1) sum += __shfl_down(sum, off, 32);
  if (lane == 0) red[wv] = sum;
  __syncthreads();
  sum = red[0] + red[1] + red[2] + red[3] + red[4] + red[5] + red[6] + red[7];
  const float inv = 1.0f / sum;
#pragma unroll
  for (int j = 0; j < 4; ++j) p[threadIdx.x + j * 256] = (f16_t)(v[j] * inv);
}

// ---------------------------------------------------------------------------
// 7) O[b] = P[b] (1024x1024) @ V[b] via vT (column-major B), f16 out.
// ---------------------------------------------------------------------------
__global__ void k_attnout(const f16_t* __restrict__ P, const f16_t* __restrict__ vT,
                          f16_t* __restrict__ O) {
  const int b = blockIdx.z;
  v8f acc[2][2] = {};
  wmma_gemm_tile(P  + (size_t)b * HW * HW + (size_t)blockIdx.x * 128 * HW, HW,
                 vT + (size_t)b * CH * HW + (size_t)blockIdx.y * 64  * HW, HW, HW, acc);
  f16_t* Ob = O + (size_t)b * HW * CH;
  int m0, n0;
  tile_origin(blockIdx.x, blockIdx.y, m0, n0);
  const int lane = threadIdx.x & 31;
#pragma unroll
  for (int mi = 0; mi < 2; ++mi)
#pragma unroll
    for (int ni = 0; ni < 2; ++ni) {
      const int n = n0 + ni * 16 + (lane & 15);
      const int mb = m0 + mi * 16 + (lane >> 4) * 8;
#pragma unroll
      for (int i = 0; i < 8; ++i)
        Ob[(size_t)(mb + i) * CH + n] = (f16_t)acc[mi][ni][i];
    }
}

// ---------------------------------------------------------------------------
// 8) Proj GEMM + bias + residual: out = x + O @ proj_w + proj_b (fp32 out).
// ---------------------------------------------------------------------------
__global__ void k_proj(const f16_t* __restrict__ O, const f16_t* __restrict__ wT,
                       const float* __restrict__ bias, const float* __restrict__ x,
                       float* __restrict__ out) {
  v8f acc[2][2] = {};
  wmma_gemm_tile(O  + (size_t)blockIdx.x * 128 * CH, CH,
                 wT + (size_t)blockIdx.y * 64  * CH, CH, CH, acc);
  int m0, n0;
  tile_origin(blockIdx.x, blockIdx.y, m0, n0);
  const int lane = threadIdx.x & 31;
#pragma unroll
  for (int mi = 0; mi < 2; ++mi)
#pragma unroll
    for (int ni = 0; ni < 2; ++ni) {
      const int n = n0 + ni * 16 + (lane & 15);
      const int mb = m0 + mi * 16 + (lane >> 4) * 8;
      const float bn = bias[n];
#pragma unroll
      for (int i = 0; i < 8; ++i) {
        const size_t idx = (size_t)(mb + i) * CH + n;
        out[idx] = x[idx] + acc[mi][ni][i] + bn;
      }
    }
}

// ---------------------------------------------------------------------------
// Host launcher. Workspace budget ~271 MB:
//   stats 2KB | qkv_wT 1.5MB | proj_wT 0.5MB | h 33.5MB | q 33.5 | k 33.5 |
//   vT 33.5 | S 134MB.  Aliases (stream-ordered, safe): P -> q..k, O -> h.
// ---------------------------------------------------------------------------
extern "C" void kernel_launch(void* const* d_in, const int* in_sizes, int n_in,
                              void* d_out, int out_size, void* d_ws, size_t ws_size,
                              hipStream_t stream) {
  (void)in_sizes; (void)n_in; (void)out_size; (void)ws_size;
  const float* x      = (const float*)d_in[0];
  const float* gamma  = (const float*)d_in[1];
  const float* beta   = (const float*)d_in[2];
  const float* qkv_w  = (const float*)d_in[3];
  const float* qkv_b  = (const float*)d_in[4];
  const float* proj_w = (const float*)d_in[5];
  const float* proj_b = (const float*)d_in[6];
  float* out = (float*)d_out;

  char* ws = (char*)d_ws;
  size_t off = 0;
  auto carve = [&](size_t bytes) -> char* {
    off = (off + 255) & ~(size_t)255;
    char* p = ws + off;
    off += bytes;
    return p;
  };
  float*  stats  = (float*)carve(NB * GRP * 2 * sizeof(float));
  f16_t*  wqkvT  = (f16_t*)carve((size_t)3 * CH * CH * sizeof(f16_t));
  f16_t*  wprojT = (f16_t*)carve((size_t)CH * CH * sizeof(f16_t));
  f16_t*  h      = (f16_t*)carve((size_t)MTOT * CH * sizeof(f16_t));
  f16_t*  qbuf   = (f16_t*)carve((size_t)MTOT * CH * sizeof(f16_t));
  f16_t*  kbuf   = (f16_t*)carve((size_t)MTOT * CH * sizeof(f16_t));
  f16_t*  vT     = (f16_t*)carve((size_t)MTOT * CH * sizeof(f16_t));
  float*  S      = (float*)carve((size_t)NB * HW * HW * sizeof(float));
  f16_t*  P = qbuf;   // alias: q,k dead after k_scores; P (67MB) spans q..k
  f16_t*  O = h;      // alias: h dead after k_qkv
  (void)kbuf;

  k_gn_stats<<<NB * GRP, 256, 0, stream>>>(x, stats);
  k_wt_cvt<<<(CH * 3 * CH + 255) / 256, 256, 0, stream>>>(qkv_w, wqkvT, CH, 3 * CH);
  k_wt_cvt<<<(CH * CH + 255) / 256, 256, 0, stream>>>(proj_w, wprojT, CH, CH);
  k_gn_apply<<<(MTOT * CH) / (256 * 4), 256, 0, stream>>>(x, gamma, beta, stats, h);
  k_qkv<<<dim3(MTOT / 128, (3 * CH) / 64), 256, 0, stream>>>(h, wqkvT, qkv_b, qbuf, kbuf, vT);
  k_scores<<<dim3(HW / 128, HW / 64, NB), 256, 0, stream>>>(qbuf, kbuf, S);
  k_softmax<<<dim3(HW, NB), 256, 0, stream>>>(S, P);
  k_attnout<<<dim3(HW / 128, CH / 64, NB), 256, 0, stream>>>(P, vT, O);
  k_proj<<<dim3(MTOT / 128, CH / 64), 256, 0, stream>>>(O, wprojT, proj_b, x, out);
}
